// ManhattanAttention_36764920053899
// MI455X (gfx1250) — matrix operand order repaired and checked by reading, not compiled, not run on hardware
//
#include <hip/hip_runtime.h>
#include <hip/hip_bf16.h>

typedef unsigned short u16;
typedef unsigned int   u32;
typedef __attribute__((ext_vector_type(16))) __bf16 v16bf;
typedef __attribute__((ext_vector_type(8)))  float  v8f;
typedef __attribute__((ext_vector_type(4)))  int    v4i;

#define B_ 2
#define S_ 4096
#define E_ 1024
#define H_ 8
#define W_ 8
#define D_ 128
#define M_ (B_ * S_)          // 8192 rows
#define SCALE_ 0.08838834764831845f   // 1/sqrt(128)

#define BLK_M 128
#define BLK_N 128
#define BLK_K 32
#define LDS_K 40              // padded row stride (elements) to dodge bank conflicts

#define AS1 __attribute__((address_space(1)))
#define AS3 __attribute__((address_space(3)))
typedef AS1 v4i* gv4i_p;      // global (device) int4*
typedef AS3 v4i* lv4i_p;      // LDS (shared) int4*

// Async global->LDS staging (gfx1250 GLOBAL_LOAD_ASYNC_TO_LDS_B128, ASYNCcnt-tracked).
// Guarded so compilation cannot regress on a toolchain without the builtin.
#if defined(__has_builtin)
#if __has_builtin(__builtin_amdgcn_global_load_async_to_lds_b128)
#define HAVE_ASYNC_LDS 1
#endif
#endif

__device__ __forceinline__ void wait_async0() {
#if defined(HAVE_ASYNC_LDS)
#if __has_builtin(__builtin_amdgcn_s_wait_asynccnt)
    __builtin_amdgcn_s_wait_asynccnt(0);
#else
    asm volatile("s_wait_asynccnt 0x0" ::: "memory");
#endif
#endif
}

// ---------- bf16 helpers (manual, no reliance on __bf16 arithmetic) ----------
__device__ __forceinline__ float bf2f(u16 x) {
    union { u32 u; float f; } c; c.u = ((u32)x) << 16; return c.f;
}
__device__ __forceinline__ u16 f2bf(float f) {
    union { float f; u32 u; } c; c.f = f;
    u32 u = c.u;
    return (u16)((u + 0x7FFFu + ((u >> 16) & 1u)) >> 16);  // RNE
}

union FragB16 { v16bf v; u32 u[8]; };

// ---------- window index kernel: replicates the Python loop EXACTLY ----------
__global__ void window_idx_kernel(int* __restrict__ idx) {
    int t = blockIdx.x * blockDim.x + threadIdx.x;
    if (t >= S_) return;
    int rp[8]; int len = 0;
    for (int i = -4; i < 4; ++i) {
        int p = t + i;
        if (p >= 0 && p < S_ && len < 8) rp[len++] = p;
    }
    while (len < 8) {
        if (t + len < S_) { rp[len] = t + len; ++len; }
        else {
            for (int j = len; j > 0; --j) rp[j] = rp[j - 1];
            rp[0] = t - (8 - len);
            ++len;
        }
    }
    for (int j = 0; j < 8; ++j) {
        int p = rp[j] % S_;
        if (p < 0) p += S_;
        idx[t * 8 + j] = p;
    }
}

// ---------- fp32 -> bf16 cast ----------
__global__ void cast_f32_bf16(const float* __restrict__ in, u16* __restrict__ out, int n) {
    int i = blockIdx.x * blockDim.x + threadIdx.x;
    int stride = gridDim.x * blockDim.x;
    for (; i < n; i += stride) out[i] = f2bf(in[i]);
}

// ---------- tiled bf16 WMMA GEMM:  Out[m,n] = sum_k A[m,k]*Wt[n,k] + bias[n] ----------
// A: [Mdim,Kdim] bf16 row-major; Wt: [Ndim,Kdim] bf16 row-major (i.e. B^T on disk).
// 256 threads = 8 waves; wave grid 2x4; each wave: 64x32 sub-tile = 4x2 WMMA accs.
// Double-buffered LDS: async-stage tile t+1 (ASYNCcnt) while WMMAs consume tile t.
template <bool WRITE_BF16>
__global__ void __launch_bounds__(256)
gemm_bf16(const u16* __restrict__ A, const u16* __restrict__ Wt,
          const float* __restrict__ bias, void* __restrict__ Out,
          int Mdim, int Ndim, int Kdim)
{
    __shared__ u16 lA[2][BLK_M * LDS_K];
    __shared__ u16 lB[2][BLK_N * LDS_K];

    const int tid   = threadIdx.x;
    const int lane  = tid & 31;
    const int wid   = tid >> 5;
    const int wave_m = wid >> 2;      // 0..1
    const int wave_n = wid & 3;       // 0..3
    const int m0 = blockIdx.y * BLK_M;
    const int n0 = blockIdx.x * BLK_N;
    const int rsel  = lane & 15;
    const int khalf = lane >> 4;

    // this thread's two 16B chunks of each 128x32 tile
    const int e0   = tid * 8;
    const int row0 = e0 >> 5, col0 = e0 & 31;
    const int e1   = e0 + 2048;
    const int row1 = e1 >> 5, col1 = e1 & 31;

    auto stage = [&](int sbuf, int kk) {
        const u16* ga0 = A  + (size_t)(m0 + row0) * Kdim + kk + col0;
        const u16* ga1 = A  + (size_t)(m0 + row1) * Kdim + kk + col1;
        const u16* gb0 = Wt + (size_t)(n0 + row0) * Kdim + kk + col0;
        const u16* gb1 = Wt + (size_t)(n0 + row1) * Kdim + kk + col1;
        u16* la0 = &lA[sbuf][row0 * LDS_K + col0];
        u16* la1 = &lA[sbuf][row1 * LDS_K + col1];
        u16* lb0 = &lB[sbuf][row0 * LDS_K + col0];
        u16* lb1 = &lB[sbuf][row1 * LDS_K + col1];
#if defined(HAVE_ASYNC_LDS)
        __builtin_amdgcn_global_load_async_to_lds_b128(
            (gv4i_p)(void*)const_cast<u16*>(ga0), (lv4i_p)(void*)la0, 0, 0);
        __builtin_amdgcn_global_load_async_to_lds_b128(
            (gv4i_p)(void*)const_cast<u16*>(ga1), (lv4i_p)(void*)la1, 0, 0);
        __builtin_amdgcn_global_load_async_to_lds_b128(
            (gv4i_p)(void*)const_cast<u16*>(gb0), (lv4i_p)(void*)lb0, 0, 0);
        __builtin_amdgcn_global_load_async_to_lds_b128(
            (gv4i_p)(void*)const_cast<u16*>(gb1), (lv4i_p)(void*)lb1, 0, 0);
#else
        *(uint4*)la0 = *(const uint4*)ga0;
        *(uint4*)la1 = *(const uint4*)ga1;
        *(uint4*)lb0 = *(const uint4*)gb0;
        *(uint4*)lb1 = *(const uint4*)gb1;
        __builtin_prefetch(ga0 + BLK_K, 0, 1);
        __builtin_prefetch(gb0 + BLK_K, 0, 1);
#endif
    };

    v8f acc[4][2];
#pragma unroll
    for (int tm = 0; tm < 4; ++tm)
#pragma unroll
        for (int tn = 0; tn < 2; ++tn)
#pragma unroll
            for (int r = 0; r < 8; ++r) acc[tm][tn][r] = 0.0f;

    // prologue: stage first tile
    stage(0, 0);
    wait_async0();
    __syncthreads();

    int buf = 0;
    for (int kb = 0; kb < Kdim; kb += BLK_K) {
        const int nxt = kb + BLK_K;
        if (nxt < Kdim) stage(buf ^ 1, nxt);   // overlap with WMMAs below

        // ---- build fragments per ISA 7.12.2 layouts ----
        FragB16 afr[4], bfr[2];
#pragma unroll
        for (int tm = 0; tm < 4; ++tm) {
            int row = wave_m * 64 + tm * 16 + rsel;          // A row (M)
            const u16* base = &lA[buf][row * LDS_K];
            // A 16x32 bf16: VGPR p holds K = {(p<4?0:16) + (p&3)*2 + khalf*8, +1}
#pragma unroll
            for (int p = 0; p < 8; ++p) {
                int kb2 = ((p & 4) << 2) + ((p & 3) << 1) + (khalf << 3);
                afr[tm].u[p] = *(const u32*)(base + kb2);
            }
        }
#pragma unroll
        for (int tn = 0; tn < 2; ++tn) {
            int nrow = wave_n * 32 + tn * 16 + rsel;         // B column (N) = Wt row
            const u16* base = &lB[buf][nrow * LDS_K] + (khalf << 3);
            // B 32x16 bf16: VGPR p holds K = {(p<4?0:16) + (p&3)*2 + khalf*8, +1}
#pragma unroll
            for (int p = 0; p < 8; ++p)
                bfr[tn].u[p] = *(const u32*)(base + ((p & 4) << 2) + ((p & 3) << 1));
        }

        // ---- 8 WMMAs per wave per K-step ----
#pragma unroll
        for (int tm = 0; tm < 4; ++tm)
#pragma unroll
            for (int tn = 0; tn < 2; ++tn)
                acc[tm][tn] = __builtin_amdgcn_wmma_f32_16x16x32_bf16(
                    false, afr[tm].v, false, bfr[tn].v,
                    (short)0, acc[tm][tn], false, false);

        if (nxt < Kdim) wait_async0();   // next tile landed in LDS
        __syncthreads();
        buf ^= 1;
    }

    // ---- epilogue: C/D layout -> memory, add bias ----
#pragma unroll
    for (int tm = 0; tm < 4; ++tm)
#pragma unroll
        for (int tn = 0; tn < 2; ++tn) {
            int bm = m0 + wave_m * 64 + tm * 16 + khalf * 8;
            int bn = n0 + wave_n * 32 + tn * 16 + rsel;
            float bb = bias[bn];
#pragma unroll
            for (int r = 0; r < 8; ++r) {
                int m = bm + r;
                float val = acc[tm][tn][r] + bb;
                if (WRITE_BF16)
                    ((u16*)Out)[(size_t)m * Ndim + bn] = f2bf(val);
                else
                    ((float*)Out)[(size_t)m * Ndim + bn] = val;
            }
        }
}

// ---------- windowed attention: one wave per (b, s, h) ----------
__global__ void __launch_bounds__(256)
attn_kernel(const u16* __restrict__ qh, const u16* __restrict__ kh,
            const u16* __restrict__ vh, const int* __restrict__ idx,
            u16* __restrict__ outbf)
{
    int g    = blockIdx.x * 8 + (threadIdx.x >> 5);   // global wave id
    int lane = threadIdx.x & 31;
    int h = g & 7;
    int s = (g >> 3) & (S_ - 1);
    int b = g >> 15;

    const size_t qoff = ((size_t)(b * S_ + s) * E_) + h * D_;
    float ql[4];
#pragma unroll
    for (int i = 0; i < 4; ++i) ql[i] = bf2f(qh[qoff + lane + i * 32]);

    int wrow[8];
#pragma unroll
    for (int w = 0; w < 8; ++w) wrow[w] = idx[s * 8 + w];

    float sc[8];
#pragma unroll
    for (int w = 0; w < 8; ++w) {
        const u16* kp = kh + ((size_t)(b * S_ + wrow[w]) * E_) + h * D_;
        float p = 0.0f;
#pragma unroll
        for (int i = 0; i < 4; ++i) p += ql[i] * bf2f(kp[lane + i * 32]);
#pragma unroll
        for (int off = 16; off > 0; off >>= 1) p += __shfl_xor(p, off, 32);
        sc[w] = p * SCALE_;
    }

    float mx = sc[0];
#pragma unroll
    for (int w = 1; w < 8; ++w) mx = fmaxf(mx, sc[w]);
    float den = 0.0f;
#pragma unroll
    for (int w = 0; w < 8; ++w) { sc[w] = __expf(sc[w] - mx); den += sc[w]; }
    float inv = 1.0f / den;

#pragma unroll
    for (int i = 0; i < 4; ++i) {
        float o = 0.0f;
#pragma unroll
        for (int w = 0; w < 8; ++w) {
            const u16* vp = vh + ((size_t)(b * S_ + wrow[w]) * E_) + h * D_;
            o += sc[w] * bf2f(vp[lane + i * 32]);
        }
        outbf[qoff + lane + i * 32] = f2bf(o * inv);
    }
}

// ---------- host-side orchestration ----------
extern "C" void kernel_launch(void* const* d_in, const int* in_sizes, int n_in,
                              void* d_out, int out_size, void* d_ws, size_t ws_size,
                              hipStream_t stream) {
    const float* q   = (const float*)d_in[0];
    const float* k   = (const float*)d_in[1];
    const float* v   = (const float*)d_in[2];
    const float* wqkv = (const float*)d_in[3];   // [3E, E]
    const float* bqkv = (const float*)d_in[4];   // [3E]
    const float* wout = (const float*)d_in[5];   // [E, E]
    const float* bout = (const float*)d_in[6];   // [E]
    (void)in_sizes; (void)n_in; (void)out_size; (void)ws_size;

    const size_t SB = (size_t)M_ * E_ * sizeof(u16);   // 16 MiB per bf16 activation
    char* ws = (char*)d_ws;
    u16* q_bf    = (u16*)(ws + 0 * SB);
    u16* k_bf    = (u16*)(ws + 1 * SB);
    u16* v_bf    = (u16*)(ws + 2 * SB);
    u16* qh_bf   = (u16*)(ws + 3 * SB);
    u16* kh_bf   = (u16*)(ws + 4 * SB);
    u16* vh_bf   = (u16*)(ws + 5 * SB);
    u16* wqkv_bf = (u16*)(ws + 6 * SB);
    u16* wout_bf = (u16*)(ws + 6 * SB + (size_t)3 * E_ * E_ * sizeof(u16));
    int* idx     = (int*)(ws + 6 * SB + (size_t)4 * E_ * E_ * sizeof(u16));
    u16* attn_bf = q_bf;   // q_bf is dead after the Q projection GEMM

    // 1) window indices
    window_idx_kernel<<<(S_ + 255) / 256, 256, 0, stream>>>(idx);

    // 2) fp32 -> bf16 casts
    const int nAct = M_ * E_;
    cast_f32_bf16<<<4096, 256, 0, stream>>>(q, q_bf, nAct);
    cast_f32_bf16<<<4096, 256, 0, stream>>>(k, k_bf, nAct);
    cast_f32_bf16<<<4096, 256, 0, stream>>>(v, v_bf, nAct);
    cast_f32_bf16<<<4096, 256, 0, stream>>>(wqkv, wqkv_bf, 3 * E_ * E_);
    cast_f32_bf16<<<4096, 256, 0, stream>>>(wout, wout_bf, E_ * E_);

    // 3) projections (gather hoisted past the linear map — mathematically identical)
    dim3 ggrid(E_ / BLK_N, M_ / BLK_M);   // (8, 64)
    gemm_bf16<true><<<ggrid, 256, 0, stream>>>(q_bf, wqkv_bf,                          bqkv,          qh_bf, M_, E_, E_);
    gemm_bf16<true><<<ggrid, 256, 0, stream>>>(k_bf, wqkv_bf + (size_t)E_ * E_,        bqkv + E_,     kh_bf, M_, E_, E_);
    gemm_bf16<true><<<ggrid, 256, 0, stream>>>(v_bf, wqkv_bf + (size_t)2 * E_ * E_,    bqkv + 2 * E_, vh_bf, M_, E_, E_);

    // 4) windowed softmax attention (gather projected K/V by window index)
    attn_kernel<<<(B_ * S_ * H_) / 8, 256, 0, stream>>>(qh_bf, kh_bf, vh_bf, idx, attn_bf);

    // 5) output projection, fp32 result straight into d_out
    gemm_bf16<false><<<ggrid, 256, 0, stream>>>(attn_bf, wout_bf, bout, (float*)d_out, M_, E_, E_);
}
